// MPNN_41180146434468
// MI455X (gfx1250) — matrix operand order repaired and checked by reading, not compiled, not run on hardware
//
#include <hip/hip_runtime.h>
#include <hip/hip_bf16.h>

typedef __bf16 bf16;
typedef __attribute__((ext_vector_type(16))) __bf16 v16bf;
typedef __attribute__((ext_vector_type(8)))  __bf16 v8bf;
typedef __attribute__((ext_vector_type(8)))  float  v8f;

__device__ __forceinline__ float bf2f(bf16 b)  { return (float)b; }
__device__ __forceinline__ bf16  f2bf(float f) { return (bf16)f; }
__device__ __forceinline__ float siluf(float v){ return v / (1.f + __expf(-v)); }

// ---------------------------------------------------------------------------
// Row-major bf16 weight blocks (z block stored o-major: col c = o*32+h so the
// GEMM emits z[n][o][h] and the edge kernel reads 64B-contiguous per lane):
//  Wc1 (64 x 1088): rows i<40: cols[o*32+h]=w1b[h,i*32+o], cols[1024+o]=b1b[i*32+o],
//                   cols[1056+o]=root0[i,o];  rows 40..63 zero (K padded 40->64)
//  Wc2 (32 x 1152): from w2b/b2b + root1 (1056..1087) + root2 (1088..1119),
//                   cols 1120..1151 zero (NC padded so CT % 4 == 0)
//  wf0p (64 x 256), wf1p (256 x 128): straight f32->bf16 copies
// ---------------------------------------------------------------------------
__global__ void k_pack(const float* __restrict__ w1b, const float* __restrict__ b1b,
                       const float* __restrict__ root0,
                       const float* __restrict__ w2b, const float* __restrict__ b2b,
                       const float* __restrict__ root1, const float* __restrict__ root2,
                       const float* __restrict__ wf0, const float* __restrict__ wf1,
                       bf16* __restrict__ Wc1, bf16* __restrict__ Wc2,
                       bf16* __restrict__ wf0p, bf16* __restrict__ wf1p)
{
    int idx = blockIdx.x * blockDim.x + threadIdx.x;
    const int n1 = 64 * 1088, n2 = 32 * 1152, n3 = 64 * 256, n4 = 256 * 128;
    if (idx < n1) {
        int i = idx / 1088, c = idx % 1088;
        float v = 0.f;
        if (i < 40) {
            if (c < 1024)      { int o = c >> 5, h = c & 31; v = w1b[h * 1280 + i * 32 + o]; }
            else if (c < 1056) { v = b1b[i * 32 + (c - 1024)]; }
            else               { v = root0[i * 32 + (c - 1056)]; }
        }
        Wc1[idx] = f2bf(v);
    } else if (idx < n1 + n2) {
        int k = idx - n1; int i = k / 1152, c = k % 1152;
        float v = 0.f;
        if (c < 1024)      { int o = c >> 5, h = c & 31; v = w2b[h * 1024 + i * 32 + o]; }
        else if (c < 1056) v = b2b[i * 32 + (c - 1024)];
        else if (c < 1088) v = root1[i * 32 + (c - 1056)];
        else if (c < 1120) v = root2[i * 32 + (c - 1088)];
        Wc2[k] = f2bf(v);
    } else if (idx < n1 + n2 + n3) {
        int k = idx - n1 - n2; wf0p[k] = f2bf(wf0[k]);
    } else if (idx < n1 + n2 + n3 + n4) {
        int k = idx - n1 - n2 - n3; wf1p[k] = f2bf(wf1[k]);
    }
}

// ---------------------------------------------------------------------------
// Swizzle a row-major (Kpad x NC) bf16 matrix into fragment-ready order:
//   Wp[((s*CT + ct)*32 + lane)*16 + j] = W[(s*32 + j + 8*(j>>3) + 8*(lane>>4)) * NC + ct*16 + (lane&15)]
// so each GEMM B fragment is one contiguous 32-byte per-lane load.
// ---------------------------------------------------------------------------
__global__ void k_fragpack(const bf16* __restrict__ W, bf16* __restrict__ Wp,
                           int Kpad, int NC)
{
    int idx = blockIdx.x * blockDim.x + threadIdx.x;
    if (idx >= Kpad * NC) return;
    int j    = idx & 15;
    int lane = (idx >> 4) & 31;
    int rest = idx >> 9;
    int CT = NC >> 4;
    int ct = rest % CT;
    int s  = rest / CT;
    int k = s * 32 + j + 8 * (j >> 3) + 8 * (lane >> 4);
    int n = ct * 16 + (lane & 15);
    Wp[idx] = W[(size_t)k * NC + n];
}

// f32 (M x K) -> bf16 (M x Kpad), zero-padded K..Kpad
__global__ void k_cvt(const float* __restrict__ A, bf16* __restrict__ Ab,
                      int M, int K, int Kpad)
{
    int idx = blockIdx.x * blockDim.x + threadIdx.x;
    if (idx >= M * Kpad) return;
    int m = idx / Kpad, k = idx % Kpad;
    Ab[idx] = f2bf(k < K ? A[(size_t)m * K + k] : 0.f);
}

// ---------------------------------------------------------------------------
// Edge MLP first layer for both nn1 and nn2 (shared read of edge_attr)
// ---------------------------------------------------------------------------
__global__ void k_edge_hidden(const float* __restrict__ ea,
                              const float* __restrict__ w1a, const float* __restrict__ b1a,
                              const float* __restrict__ w2a, const float* __restrict__ b2a,
                              bf16* __restrict__ hid1, bf16* __restrict__ hid2, int E)
{
    int idx = blockIdx.x * blockDim.x + threadIdx.x;
    int e = idx >> 5, c = idx & 31;
    if (e >= E) return;
    const float* a = ea + (size_t)e * 10;
    float s1 = b1a[c], s2 = b2a[c];
#pragma unroll
    for (int i = 0; i < 10; ++i) {
        float av = a[i];
        s1 = fmaf(av, w1a[i * 32 + c], s1);
        s2 = fmaf(av, w2a[i * 32 + c], s2);
    }
    hid1[idx] = f2bf(fmaxf(s1, 0.f));
    hid2[idx] = f2bf(fmaxf(s2, 0.f));
}

// ---------------------------------------------------------------------------
// WMMA GEMM: C[M x NC] bf16 = A[M x Kpad] (bf16 padded) @ Wp (fragment-packed)
// Requirements (guaranteed by caller): M % 16 == 0, NC % 64 == 0, Kpad % 32 == 0.
// Wave = 16 rows x 64 cols strip, 4 unconditional WMMAs per k-step.
// rt/cg derived from a readfirstlane'd wave index -> scalar branches only.
// mode 0: raw store.  mode 1: +bias then SiLU.
// ---------------------------------------------------------------------------
__global__ void __launch_bounds__(256)
k_gemm(const bf16* __restrict__ A, const bf16* __restrict__ Wp,
       bf16* __restrict__ C, const float* __restrict__ bias,
       int M, int Kpad, int NC, int mode)
{
    int lane = threadIdx.x & 31;
    int widx = __builtin_amdgcn_readfirstlane((int)(threadIdx.x >> 5));
    int wv   = blockIdx.x * 8 + widx;
    int RT = M >> 4;
    int CT = NC >> 4;
    int CG = CT >> 2;
    int rt = wv % RT, cg = wv / RT;     // consecutive waves share cg -> B in L0
    if (cg >= CG) return;               // scalar branch
    int KS = Kpad >> 5;
    int lhi = lane >> 4, llo = lane & 15;

    const bf16* arow = A + (size_t)(rt * 16 + llo) * Kpad + 8 * lhi;
    const bf16* wcol = Wp + ((size_t)cg * 4 * 32 + lane) * 16;

    v8f acc[4] = {};
    for (int s = 0; s < KS; ++s) {
        // A 16x32 16-bit map: two contiguous 8-element runs per lane
        const bf16* ak = arow + s * 32;
        v8bf alo = *(const v8bf*)(ak);
        v8bf ahi = *(const v8bf*)(ak + 16);
        v16bf a = __builtin_shufflevector(alo, ahi, 0, 1, 2, 3, 4, 5, 6, 7,
                                          8, 9, 10, 11, 12, 13, 14, 15);
        const bf16* wb = wcol + (size_t)s * CT * 512;
        v16bf b0 = *(const v16bf*)(wb);
        v16bf b1 = *(const v16bf*)(wb + 512);
        v16bf b2 = *(const v16bf*)(wb + 1024);
        v16bf b3 = *(const v16bf*)(wb + 1536);
        acc[0] = __builtin_amdgcn_wmma_f32_16x16x32_bf16(false, a, false, b0,
                                                         (short)0, acc[0], false, false);
        acc[1] = __builtin_amdgcn_wmma_f32_16x16x32_bf16(false, a, false, b1,
                                                         (short)0, acc[1], false, false);
        acc[2] = __builtin_amdgcn_wmma_f32_16x16x32_bf16(false, a, false, b2,
                                                         (short)0, acc[2], false, false);
        acc[3] = __builtin_amdgcn_wmma_f32_16x16x32_bf16(false, a, false, b3,
                                                         (short)0, acc[3], false, false);
    }

    int n0 = cg * 64 + llo;
#pragma unroll
    for (int t = 0; t < 4; ++t) {
        int n = n0 + t * 16;
        float bv = mode ? bias[n] : 0.f;
#pragma unroll
        for (int r = 0; r < 8; ++r) {
            int m = rt * 16 + r + 8 * lhi;   // C map: VGPR r -> M=r (+8 lanes 16..31)
            float v = acc[t][r];
            if (mode) v = siluf(v + bv);
            C[(size_t)m * NC + n] = f2bf(v);
        }
    }
}

// ---------------------------------------------------------------------------
// Per-edge message + scatter-add. One wave per edge, lane = output channel o.
//   msg[o] = zb[src,o] + sum_h hid[e,h] * z[src,o,h]
// z is o-major: each lane reads 64 contiguous bytes (4x b128) out of L2.
// ---------------------------------------------------------------------------
__global__ void k_edge_msg(const bf16* __restrict__ hid, const bf16* __restrict__ zfull,
                           int NCpad, const int* __restrict__ src, const int* __restrict__ dst,
                           float* __restrict__ agg, int E)
{
    int wv = (blockIdx.x * blockDim.x + threadIdx.x) >> 5;
    int lane = threadIdx.x & 31;
    if (wv >= E) return;
    int s = src[wv], d = dst[wv];
    const bf16* zrow = zfull + (size_t)s * NCpad;
    const bf16* zl = zrow + lane * 32;
    __builtin_prefetch(zl, 0, 1);
    v8bf z0 = *(const v8bf*)(zl);
    v8bf z1 = *(const v8bf*)(zl + 8);
    v8bf z2 = *(const v8bf*)(zl + 16);
    v8bf z3 = *(const v8bf*)(zl + 24);
    float hv  = bf2f(hid[(size_t)wv * 32 + lane]);
    float acc = bf2f(zrow[1024 + lane]);            // zb (edge-bias term)
#pragma unroll
    for (int h = 0; h < 8; ++h) acc = fmaf(__shfl(hv, h, 32),      bf2f(z0[h]), acc);
#pragma unroll
    for (int h = 0; h < 8; ++h) acc = fmaf(__shfl(hv, h + 8, 32),  bf2f(z1[h]), acc);
#pragma unroll
    for (int h = 0; h < 8; ++h) acc = fmaf(__shfl(hv, h + 16, 32), bf2f(z2[h]), acc);
#pragma unroll
    for (int h = 0; h < 8; ++h) acc = fmaf(__shfl(hv, h + 24, 32), bf2f(z3[h]), acc);
    atomicAdd(&agg[(size_t)d * 32 + lane], acc);
}

__global__ void k_zero(float* __restrict__ p, int n)
{
    int i = blockIdx.x * blockDim.x + threadIdx.x;
    if (i < n) p[i] = 0.f;
}

// out[n,c] = agg[n,c] + zfull[n, rootOff+c] + bias[c]  (opt SiLU); bf16 out
__global__ void k_combine(bf16* __restrict__ out, const float* __restrict__ agg,
                          const bf16* __restrict__ zfull, int NCpad, int rootOff,
                          const float* __restrict__ bias, int N, int doSilu)
{
    int idx = blockIdx.x * blockDim.x + threadIdx.x;
    int n = idx >> 5, c = idx & 31;
    if (n >= N) return;
    float v = agg[idx] + bf2f(zfull[(size_t)n * NCpad + rootOff + c]) + bias[c];
    if (doSilu) v = siluf(v);
    out[idx] = f2bf(v);
}

// channel sums + sumsq over N rows of (N x 32) bf16 -> stats[0..31]+[32..63]
__global__ void k_stats(const bf16* __restrict__ X, float* __restrict__ stats, int N)
{
    int c = threadIdx.x & 31;
    int row = blockIdx.x * (blockDim.x >> 5) + (threadIdx.x >> 5);
    int stride = gridDim.x * (blockDim.x >> 5);
    float s = 0.f, q = 0.f;
    for (int n = row; n < N; n += stride) {
        float v = bf2f(X[(size_t)n * 32 + c]);
        s += v; q += v * v;
    }
    atomicAdd(&stats[c], s);
    atomicAdd(&stats[32 + c], q);
}

// h[n,c] = silu( (x-mean)*rsqrt(var+eps)*g + b ); bf16 in/out
__global__ void k_bn_silu(const bf16* __restrict__ X, const float* __restrict__ stats,
                          const float* __restrict__ g, const float* __restrict__ b,
                          bf16* __restrict__ out, int N)
{
    int idx = blockIdx.x * blockDim.x + threadIdx.x;
    int n = idx >> 5, c = idx & 31;
    if (n >= N) return;
    float invN = 1.f / (float)N;
    float m  = stats[c] * invN;
    float va = stats[32 + c] * invN - m * m;
    float v  = (bf2f(X[idx]) - m) * rsqrtf(va + 1e-5f) * g[c] + b[c];
    out[idx] = f2bf(siluf(v));
}

// global_add_pool: hg[batch[n], c] += h[n, c]   (hg row stride 64; cols 0..31)
__global__ void k_pool(const bf16* __restrict__ h, const int* __restrict__ batch,
                       float* __restrict__ hg, int N)
{
    int idx = blockIdx.x * blockDim.x + threadIdx.x;
    int n = idx >> 5, c = idx & 31;
    if (n >= N) return;
    int g = batch[n];
    atomicAdd(&hg[(size_t)g * 64 + c], bf2f(h[idx]));
}

// t[g,o] = mol[g,:200] @ wm0[:,o] + bm0[o]
__global__ void k_mol_a(const float* __restrict__ mol, const float* __restrict__ wm0,
                        const float* __restrict__ bm0, float* __restrict__ t, int G)
{
    int idx = blockIdx.x * blockDim.x + threadIdx.x;
    int g = idx >> 5, o = idx & 31;
    if (g >= G) return;
    const float* mrow = mol + (size_t)g * 200;
    float acc = bm0[o];
    for (int i = 0; i < 200; ++i) acc = fmaf(mrow[i], wm0[i * 32 + o], acc);
    t[idx] = acc;
}

// h_m[g,o] = silu( silu(bn(t[g,:])) @ wm1[:,o] + bm1[o] )  -> hg cols 32..63
__global__ void k_mol_b(const float* __restrict__ t, const float* __restrict__ stats,
                        const float* __restrict__ bng, const float* __restrict__ bnb,
                        const float* __restrict__ wm1, const float* __restrict__ bm1,
                        float* __restrict__ hg, int G)
{
    int idx = blockIdx.x * blockDim.x + threadIdx.x;
    int g = idx >> 5, o = idx & 31;
    if (g >= G) return;
    float invN = 1.f / (float)G;
    float acc = bm1[o];
#pragma unroll
    for (int i = 0; i < 32; ++i) {
        float m  = stats[i] * invN;
        float va = stats[32 + i] * invN - m * m;
        float v  = (t[(size_t)g * 32 + i] - m) * rsqrtf(va + 1e-5f) * bng[i] + bnb[i];
        acc = fmaf(siluf(v), wm1[i * 32 + o], acc);
    }
    hg[(size_t)g * 64 + 32 + o] = siluf(acc);
}

// out[g] = f1[g,:128] @ wf2 + bf2 ; one wave per graph, shuffle reduction
__global__ void k_head_final(const bf16* __restrict__ f1, const float* __restrict__ wf2,
                             const float* __restrict__ bf2v, float* __restrict__ out, int G)
{
    int idx = blockIdx.x * blockDim.x + threadIdx.x;
    int g = idx >> 5, lane = idx & 31;
    if (g >= G) return;
    float acc = 0.f;
#pragma unroll
    for (int j = 0; j < 4; ++j)
        acc = fmaf(bf2f(f1[(size_t)g * 128 + lane + 32 * j]), wf2[lane + 32 * j], acc);
#pragma unroll
    for (int offd = 16; offd > 0; offd >>= 1) acc += __shfl_down(acc, offd, 32);
    if (lane == 0) out[g] = acc + bf2v[0];
}

// ---------------------------------------------------------------------------
static inline void launch_gemm(const bf16* A, const bf16* Wp, bf16* C, const float* bias,
                               int M, int Kpad, int NC, int mode, hipStream_t s)
{
    int RT = M / 16, CT = NC / 16, CG = CT / 4;
    long long waves = (long long)RT * CG;
    int blocks = (int)((waves + 7) / 8);
    k_gemm<<<blocks, 256, 0, s>>>(A, Wp, C, bias, M, Kpad, NC, mode);
}

extern "C" void kernel_launch(void* const* d_in, const int* in_sizes, int n_in,
                              void* d_out, int out_size, void* d_ws, size_t ws_size,
                              hipStream_t stream)
{
    const float* x     = (const float*)d_in[0];
    const float* ea    = (const float*)d_in[1];
    const float* mol   = (const float*)d_in[2];
    const float* w1a   = (const float*)d_in[3];
    const float* b1a   = (const float*)d_in[4];
    const float* w1b   = (const float*)d_in[5];
    const float* b1b   = (const float*)d_in[6];
    const float* w2a   = (const float*)d_in[7];
    const float* b2a   = (const float*)d_in[8];
    const float* w2b   = (const float*)d_in[9];
    const float* b2b   = (const float*)d_in[10];
    const float* root0 = (const float*)d_in[11];
    const float* bias0 = (const float*)d_in[12];
    const float* root1 = (const float*)d_in[13];
    const float* bias1 = (const float*)d_in[14];
    const float* root2 = (const float*)d_in[15];
    const float* bias2 = (const float*)d_in[16];
    const float* bng   = (const float*)d_in[17];
    const float* bnb   = (const float*)d_in[18];
    const float* wm0   = (const float*)d_in[19];
    const float* bm0   = (const float*)d_in[20];
    const float* wm1   = (const float*)d_in[21];
    const float* bm1   = (const float*)d_in[22];
    const float* bnmg  = (const float*)d_in[23];
    const float* bnmb  = (const float*)d_in[24];
    const float* wf0   = (const float*)d_in[25];
    const float* bf0   = (const float*)d_in[26];
    const float* wf1   = (const float*)d_in[27];
    const float* bf1   = (const float*)d_in[28];
    const float* wf2   = (const float*)d_in[29];
    const float* bf2v  = (const float*)d_in[30];
    const int*   eidx  = (const int*)d_in[31];
    const int*   batch = (const int*)d_in[32];
    (void)n_in; (void)ws_size; (void)out_size;

    const int N = in_sizes[0] / 40;
    const int E = in_sizes[1] / 10;
    const int G = in_sizes[2] / 200;
    const int* srcI = eidx;
    const int* dstI = eidx + E;

    char* ws = (char*)d_ws;
    size_t off = 0;
    auto alloc = [&](size_t bytes) {
        off = (off + 255) & ~(size_t)255;
        size_t o = off; off += bytes; return o;
    };
    bf16*  hid1  = (bf16*)(ws + alloc((size_t)E * 32 * 2));
    bf16*  hid2  = (bf16*)(ws + alloc((size_t)E * 32 * 2));
    bf16*  Wc1   = (bf16*)(ws + alloc((size_t)64 * 1088 * 2));
    bf16*  Wc2   = (bf16*)(ws + alloc((size_t)32 * 1152 * 2));
    bf16*  wf0p  = (bf16*)(ws + alloc((size_t)64 * 256 * 2));
    bf16*  wf1p  = (bf16*)(ws + alloc((size_t)256 * 128 * 2));
    bf16*  Wc1p  = (bf16*)(ws + alloc((size_t)64 * 1088 * 2));
    bf16*  Wc2p  = (bf16*)(ws + alloc((size_t)32 * 1152 * 2));
    bf16*  wf0pp = (bf16*)(ws + alloc((size_t)64 * 256 * 2));
    bf16*  wf1pp = (bf16*)(ws + alloc((size_t)256 * 128 * 2));
    bf16*  xb    = (bf16*)(ws + alloc((size_t)N * 64 * 2));
    bf16*  zfull = (bf16*)(ws + alloc((size_t)N * 1152 * 2));
    float* agg   = (float*)(ws + alloc((size_t)N * 32 * 4));
    bf16*  hpre  = (bf16*)(ws + alloc((size_t)N * 32 * 2));
    bf16*  h0b   = (bf16*)(ws + alloc((size_t)N * 32 * 2));
    bf16*  h1b   = (bf16*)(ws + alloc((size_t)N * 32 * 2));
    bf16*  h2b   = (bf16*)(ws + alloc((size_t)N * 32 * 2));
    float* stats = (float*)(ws + alloc((size_t)64 * 4));
    float* hg    = (float*)(ws + alloc((size_t)G * 64 * 4));
    bf16*  hgb   = (bf16*)(ws + alloc((size_t)G * 64 * 2));
    float* tmol  = (float*)(ws + alloc((size_t)G * 32 * 4));
    bf16*  tmolb = (bf16*)(ws + alloc((size_t)G * 32 * 2));
    bf16*  f0b   = (bf16*)(ws + alloc((size_t)G * 256 * 2));
    bf16*  f1b   = (bf16*)(ws + alloc((size_t)G * 128 * 2));

    const int TB = 256;
    auto blk = [](long long n) { return (int)((n + 255) / 256); };

    // weight packing -> fragment swizzle -> A conversion -> edge hidden
    k_pack<<<blk(64 * 1088 + 32 * 1152 + 64 * 256 + 256 * 128), TB, 0, stream>>>(
        w1b, b1b, root0, w2b, b2b, root1, root2, wf0, wf1, Wc1, Wc2, wf0p, wf1p);
    k_fragpack<<<blk(64 * 1088), TB, 0, stream>>>(Wc1, Wc1p, 64, 1088);
    k_fragpack<<<blk(32 * 1152), TB, 0, stream>>>(Wc2, Wc2p, 32, 1152);
    k_fragpack<<<blk(64 * 256), TB, 0, stream>>>(wf0p, wf0pp, 64, 256);
    k_fragpack<<<blk(256 * 128), TB, 0, stream>>>(wf1p, wf1pp, 256, 128);
    k_cvt<<<blk((long long)N * 64), TB, 0, stream>>>(x, xb, N, 40, 64);
    k_edge_hidden<<<blk((long long)E * 32), TB, 0, stream>>>(ea, w1a, b1a, w2a, b2a,
                                                             hid1, hid2, E);

    // ----- layer 0: z/zb/root GEMM (K 40->64), messages, BN+SiLU -----
    launch_gemm(xb, Wc1p, zfull, nullptr, N, 64, 1088, 0, stream);
    k_zero<<<blk((long long)N * 32), TB, 0, stream>>>(agg, N * 32);
    k_edge_msg<<<blk((long long)E * 32), TB, 0, stream>>>(hid1, zfull, 1088, srcI, dstI, agg, E);
    k_combine<<<blk((long long)N * 32), TB, 0, stream>>>(hpre, agg, zfull, 1088, 1056,
                                                         bias0, N, 0);
    k_zero<<<1, 64, 0, stream>>>(stats, 64);
    k_stats<<<512, TB, 0, stream>>>(hpre, stats, N);
    k_bn_silu<<<blk((long long)N * 32), TB, 0, stream>>>(hpre, stats, bng, bnb, h0b, N);

    // ----- layer 1 (shared edge MLP nn2, root1) -----
    launch_gemm(h0b, Wc2p, zfull, nullptr, N, 32, 1152, 0, stream);
    k_zero<<<blk((long long)N * 32), TB, 0, stream>>>(agg, N * 32);
    k_edge_msg<<<blk((long long)E * 32), TB, 0, stream>>>(hid2, zfull, 1152, srcI, dstI, agg, E);
    k_combine<<<blk((long long)N * 32), TB, 0, stream>>>(h1b, agg, zfull, 1152, 1056,
                                                         bias1, N, 1);

    // ----- layer 2 (shared edge MLP nn2, root2) -----
    launch_gemm(h1b, Wc2p, zfull, nullptr, N, 32, 1152, 0, stream);
    k_zero<<<blk((long long)N * 32), TB, 0, stream>>>(agg, N * 32);
    k_edge_msg<<<blk((long long)E * 32), TB, 0, stream>>>(hid2, zfull, 1152, srcI, dstI, agg, E);
    k_combine<<<blk((long long)N * 32), TB, 0, stream>>>(h2b, agg, zfull, 1152, 1088,
                                                         bias2, N, 1);

    // ----- pooling + molecular-feature MLP into concat buffer hg (G x 64) -----
    k_zero<<<blk((long long)G * 64), TB, 0, stream>>>(hg, G * 64);
    k_pool<<<blk((long long)N * 32), TB, 0, stream>>>(h2b, batch, hg, N);
    k_mol_a<<<blk((long long)G * 32), TB, 0, stream>>>(mol, wm0, bm0, tmol, G);
    k_zero<<<1, 64, 0, stream>>>(stats, 64);
    k_cvt<<<blk((long long)G * 32), TB, 0, stream>>>(tmol, tmolb, G, 32, 32);
    k_stats<<<128, TB, 0, stream>>>(tmolb, stats, G);
    k_mol_b<<<blk((long long)G * 32), TB, 0, stream>>>(tmol, stats, bnmg, bnmb, wm1, bm1, hg, G);
    k_cvt<<<blk((long long)G * 64), TB, 0, stream>>>(hg, hgb, G, 64, 64);

    // ----- head: two WMMA GEMMs with fused bias+SiLU, then final dot -----
    launch_gemm(hgb, wf0pp, f0b, bf0, G, 64, 256, 1, stream);
    launch_gemm(f0b, wf1pp, f1b, bf1, G, 256, 128, 1, stream);
    k_head_final<<<blk((long long)G * 32), TB, 0, stream>>>(f1b, wf2, bf2v, (float*)d_out, G);
}
